// Transformer_73254962200937
// MI455X (gfx1250) — matrix operand order repaired and checked by reading, compiled, not run on hardware
//
#include <hip/hip_runtime.h>
#include <hip/hip_bf16.h>
#include <math.h>

// ---------------------------------------------------------------------------
// 6-layer transformer forward for MI455X (gfx1250, wave32, WMMA).
// Compute-bound (~425 GFLOP vs ~250MB HBM) => bf16 WMMA f32-accumulate for all
// GEMMs. Weights are pre-converted+transposed to bf16 [N][K] once per launch
// (~10us of HBM time) so GEMM tile staging is a pure 16B copy -> eligible for
// the CDNA5 async global->LDS path (GLOBAL_LOAD_ASYNC_TO_LDS_B128, ASYNCcnt)
// with LDS double buffering. Falls back to manual copies if the builtin is
// not available in this toolchain.
// ---------------------------------------------------------------------------

#define RTOK   4096   // 4 * 1024 token rows
#define DIMM   768
#define NHEAD  12
#define DHEAD  64
#define MLPD   3072
#define SEQ    1024
#define NBATCH 4
#define NLAYER 6

#if defined(__HIP_DEVICE_COMPILE__) && __has_builtin(__builtin_amdgcn_global_load_async_to_lds_b128)
#define ATH_ASYNC 1
#else
#define ATH_ASYNC 0
#endif

typedef __attribute__((ext_vector_type(16))) __bf16 v16bf;
typedef __attribute__((ext_vector_type(8)))  float  v8f;

union AFu {                 // one WMMA 16-bit A/B fragment (16 bf16 = 32B)
    v16bf v;
    uint4 u4[2];
    unsigned short s[16];
};

__device__ __forceinline__ unsigned short f2bf_u(float f) {
    union { float f; unsigned u; } a; a.f = f;
    unsigned u = a.u;
    return (unsigned short)((u + 0x7fffu + ((u >> 16) & 1u)) >> 16); // RNE
}
__device__ __forceinline__ __bf16 f2bf(float f) {
    union { unsigned short s; __bf16 b; } o; o.s = f2bf_u(f); return o.b;
}

// 16-byte global->LDS copy: async (no VGPR round trip, ASYNCcnt) if available.
// Builtin signature (from probe diagnostic): params are pointers to
// int __attribute__((vector_size(16))) with global / LDS address spaces.
typedef int v4i_gcc __attribute__((vector_size(16)));
typedef __attribute__((address_space(1))) v4i_gcc* g_v4i_ptr;
typedef __attribute__((address_space(3))) v4i_gcc* l_v4i_ptr;

__device__ __forceinline__ void cp16_g2s(void* l, const void* g) {
#if ATH_ASYNC
    __builtin_amdgcn_global_load_async_to_lds_b128((g_v4i_ptr)g, (l_v4i_ptr)l, 0, 0);
#else
    *reinterpret_cast<uint4*>(l) = *reinterpret_cast<const uint4*>(g);
#endif
}
__device__ __forceinline__ void async_wait0() {
#if ATH_ASYNC
#if __has_builtin(__builtin_amdgcn_s_wait_asynccnt)
    __builtin_amdgcn_s_wait_asynccnt(0);
#else
    asm volatile("s_wait_asynccnt 0x0" ::: "memory");
#endif
#endif
}

// ---------------------------------------------------------------------------
// One-time weight prep: W [K x N] f32  ->  Wt [N x K] bf16 (transposed), so
// GEMM B-tiles are contiguous bf16 rows. Tiled through LDS, both sides
// coalesced. gridDim.z = layer.
// ---------------------------------------------------------------------------
__global__ __launch_bounds__(256) void wconv_kernel(const float* __restrict__ W,
                                                    __bf16* __restrict__ Wt,
                                                    int K, int N) {
    __shared__ float t[32][33];
    const int n0 = blockIdx.x * 32, k0 = blockIdx.y * 32;
    const size_t lo = (size_t)blockIdx.z * K * N;
    const int tx = threadIdx.x & 31, ty = threadIdx.x >> 5;
#pragma unroll
    for (int j = 0; j < 4; ++j) {
        int kk = ty + j * 8;
        t[kk][tx] = W[lo + (size_t)(k0 + kk) * N + n0 + tx];
    }
    __syncthreads();
#pragma unroll
    for (int j = 0; j < 4; ++j) {
        int nn = ty + j * 8;
        Wt[lo + (size_t)(n0 + nn) * K + k0 + tx] = f2bf(t[tx][nn]);
    }
}

// ---------------------------------------------------------------------------
// LayerNorm: one wave per 768-wide row, bf16 output.
// ---------------------------------------------------------------------------
__global__ __launch_bounds__(256) void ln_kernel(const float* __restrict__ x,
                                                 const float* __restrict__ w,
                                                 const float* __restrict__ b,
                                                 __bf16* __restrict__ out) {
    const int lane = threadIdx.x & 31, wave = threadIdx.x >> 5;
    const int row  = blockIdx.x * 8 + wave;
    const float* xr = x + (size_t)row * DIMM;
    float vals[24], s = 0.f, sq = 0.f;
#pragma unroll
    for (int j = 0; j < 24; ++j) {
        float v = xr[lane + 32 * j];
        vals[j] = v; s += v; sq += v * v;
    }
#pragma unroll
    for (int o = 16; o > 0; o >>= 1) {
        s  += __shfl_xor(s,  o, 32);
        sq += __shfl_xor(sq, o, 32);
    }
    const float mean = s * (1.f / DIMM);
    const float var  = sq * (1.f / DIMM) - mean * mean;
    const float rstd = rsqrtf(var + 1e-5f);
    __bf16* orow = out + (size_t)row * DIMM;
#pragma unroll
    for (int j = 0; j < 24; ++j) {
        int c = lane + 32 * j;
        orow[c] = f2bf((vals[j] - mean) * rstd * w[c] + b[c]);
    }
}

// ---------------------------------------------------------------------------
// bf16 WMMA GEMM: C[4096 x N] = A[4096 x K](bf16, row-major)
//                              * Wt[N x K](bf16, pre-transposed)^T.
// Block 256 thr (8 waves), tile 128x128x32, wave grid 2x4, wave tile 64x32
// (4x2 frags -> 8 WMMA per K-step). LDS double-buffered; tiles staged with
// async global->LDS copies overlapped with WMMA on the other buffer.
// LDS rows are 56 elems (112B): 16B-aligned for b128 ops and stride 28 dwords
// => 16 consecutive rows touch 16 distinct banks (conflict-free frag loads).
// EPI 0: split->q/k/v bf16 (q scaled)   EPI 1: +bias +residual -> f32
// EPI 2: +bias, exact GELU -> bf16
// ---------------------------------------------------------------------------
#define LDG 56

template <int EPI>
__global__ __launch_bounds__(256) void gemm_bf16_kernel(
    const __bf16* __restrict__ A, const __bf16* __restrict__ Wt,
    const float* __restrict__ bias, int K, int N, float qscale,
    __bf16* __restrict__ oQ, __bf16* __restrict__ oK, __bf16* __restrict__ oV,
    float* __restrict__ oF, __bf16* __restrict__ oB) {
    __shared__ __align__(16) __bf16 As[2][128 * LDG];
    __shared__ __align__(16) __bf16 Bs[2][128 * LDG];

    const int tid = threadIdx.x, lane = tid & 31, wave = tid >> 5;
    const int wm = wave >> 2, wn = wave & 3;            // 2 x 4 wave grid
    const int m0 = blockIdx.y * 128, n0 = blockIdx.x * 128;
    const int halfsel = lane >> 4, nloc = lane & 15;

    const __bf16* Ag = A  + (size_t)m0 * K;
    const __bf16* Bg = Wt + (size_t)n0 * K;

    // stage one 128x32 A-tile + 128x32 B-tile (512 16B chunks each -> 2/thread)
    auto stage = [&](int buf, int k0) {
#pragma unroll
        for (int i = 0; i < 2; ++i) {
            const int c = tid + i * 256;
            const int row = c >> 2, c8 = (c & 3) * 8;
            cp16_g2s(&As[buf][row * LDG + c8], Ag + (size_t)row * K + k0 + c8);
            cp16_g2s(&Bs[buf][row * LDG + c8], Bg + (size_t)row * K + k0 + c8);
        }
    };

    const v8f vz = {0.f, 0.f, 0.f, 0.f, 0.f, 0.f, 0.f, 0.f};
    v8f acc[4][2];
#pragma unroll
    for (int i = 0; i < 4; ++i)
#pragma unroll
        for (int j = 0; j < 2; ++j) acc[i][j] = vz;

    const int nk = K / 32;
    stage(0, 0);
    for (int kk = 0; kk < nk; ++kk) {
        const int cur = kk & 1;
        async_wait0();          // tile[cur] copies done (this wave)
        __syncthreads();        // ... and every other wave's too

        if constexpr (ATH_ASYNC != 0) {     // async: kick next tile ASAP
            if (kk + 1 < nk) stage(cur ^ 1, (kk + 1) * 32);
        }

        AFu a[4], b[2];
#pragma unroll
        for (int mi = 0; mi < 4; ++mi) {    // ISA 16-bit A layout
            const __bf16* p = &As[cur][(wm * 64 + mi * 16 + nloc) * LDG + halfsel * 8];
            a[mi].u4[0] = *reinterpret_cast<const uint4*>(p);
            a[mi].u4[1] = *reinterpret_cast<const uint4*>(p + 16);
        }
#pragma unroll
        for (int ni = 0; ni < 2; ++ni) {    // ISA 16-bit B layout
            const __bf16* p = &Bs[cur][(wn * 32 + ni * 16 + nloc) * LDG + halfsel * 8];
            b[ni].u4[0] = *reinterpret_cast<const uint4*>(p);
            b[ni].u4[1] = *reinterpret_cast<const uint4*>(p + 16);
        }

        if constexpr (ATH_ASYNC == 0) {     // fallback: stage after frag loads
            if (kk + 1 < nk) stage(cur ^ 1, (kk + 1) * 32);
        }

#pragma unroll
        for (int mi = 0; mi < 4; ++mi)
#pragma unroll
            for (int ni = 0; ni < 2; ++ni)
                acc[mi][ni] = __builtin_amdgcn_wmma_f32_16x16x32_bf16(
                    false, a[mi].v, false, b[ni].v, (short)0, acc[mi][ni], false, false);
    }

    // ---- epilogue ----
#pragma unroll
    for (int mi = 0; mi < 4; ++mi)
#pragma unroll
        for (int ni = 0; ni < 2; ++ni) {
            const int colg = n0 + wn * 32 + ni * 16 + nloc;
#pragma unroll
            for (int r = 0; r < 8; ++r) {
                const int rowg = m0 + wm * 64 + mi * 16 + r + 8 * halfsel;
                float vacc = acc[mi][ni][r];
                if constexpr (EPI == 0) {
                    const int sec = n0 / DIMM;           // uniform per block
                    const int cw  = colg - sec * DIMM;
                    __bf16* dst = (sec == 0) ? oQ : ((sec == 1) ? oK : oV);
                    const float sc = (sec == 0) ? qscale : 1.0f;
                    dst[(size_t)rowg * DIMM + cw] = f2bf(vacc * sc);
                } else if constexpr (EPI == 1) {
                    size_t idx = (size_t)rowg * N + colg;
                    oF[idx] = vacc + bias[colg] + oF[idx];
                } else {
                    float xg = vacc + bias[colg];
                    float g  = 0.5f * xg * (1.0f + erff(xg * 0.70710678118654752f));
                    oB[(size_t)rowg * N + colg] = f2bf(g);
                }
            }
        }
}

// ---------------------------------------------------------------------------
// Flash attention: block = (batch,head,64-query tile), 128 thr = 4 waves,
// each wave owns 16 query rows. Streams 64-key blocks; WMMA for S=Q*K^T and
// O+=P*V; online softmax via lane-half shuffles (matches C-layout rows).
// K tile staged with async global->LDS copies; V transposed manually.
// ---------------------------------------------------------------------------
#define LDT 72   // 144-byte rows: every uint4 access naturally aligned

__global__ __launch_bounds__(128) void attn_kernel(const __bf16* __restrict__ q,
                                                   const __bf16* __restrict__ k,
                                                   const __bf16* __restrict__ v,
                                                   __bf16* __restrict__ o) {
    __shared__ __align__(16) __bf16 Ks[64 * LDT];       // [key][feat]
    __shared__ __align__(16) __bf16 Vt[64 * LDT];       // [feat][key]
    __shared__ __align__(16) __bf16 Ps[4 * 16 * LDT];   // per-wave P staging

    const int tid = threadIdx.x, lane = tid & 31, wave = tid >> 5;
    const int qt = blockIdx.x, h = blockIdx.y, bz = blockIdx.z;
    const int half = lane >> 4, nloc = lane & 15;
    const size_t rowbase = (size_t)bz * SEQ;
    const int hc = h * DHEAD;

    const v8f vz = {0.f, 0.f, 0.f, 0.f, 0.f, 0.f, 0.f, 0.f};

    // Q fragments (scale already folded into q): 2 k-steps over head dim 64
    AFu qa[2];
    {
        const size_t qrow = rowbase + qt * 64 + wave * 16 + nloc;
        const __bf16* qp = q + qrow * DIMM + hc;
#pragma unroll
        for (int s = 0; s < 2; ++s) {
            int kb = s * 32 + half * 8;
            qa[s].u4[0] = *reinterpret_cast<const uint4*>(qp + kb);
            qa[s].u4[1] = *reinterpret_cast<const uint4*>(qp + kb + 16);
        }
    }

    v8f Oacc[4];
#pragma unroll
    for (int nt = 0; nt < 4; ++nt) Oacc[nt] = vz;
    float mrow[8], lrow[8];
#pragma unroll
    for (int r = 0; r < 8; ++r) { mrow[r] = -1e30f; lrow[r] = 0.f; }

    __bf16* Pw = Ps + wave * 16 * LDT;

    for (int kb0 = 0; kb0 < SEQ; kb0 += 64) {
        // ---- stage K tile [64 keys][64 feats] (async path if available) ----
        {
            int r = tid >> 1, seg = tid & 1;
            const __bf16* src = k + (rowbase + kb0 + r) * DIMM + hc + seg * 32;
            __bf16* dst = &Ks[r * LDT + seg * 32];
#pragma unroll
            for (int j = 0; j < 4; ++j) cp16_g2s(dst + j * 8, src + j * 8);
        }
        // ---- stage V tile transposed [64 feats][64 keys] ----
#pragma unroll
        for (int j = 0; j < 4; ++j) {
            int cc = tid + 128 * j;
            int key = cc >> 3, f8 = (cc & 7) * 8;
            union { uint4 u; unsigned short s[8]; } t;
            t.u = *reinterpret_cast<const uint4*>(v + (rowbase + kb0 + key) * DIMM + hc + f8);
#pragma unroll
            for (int i = 0; i < 8; ++i) {
                union { unsigned short s; __bf16 b; } cv; cv.s = t.s[i];
                Vt[(f8 + i) * LDT + key] = cv.b;
            }
        }
        async_wait0();
        __syncthreads();

        // ---- S = Q * K^T  (16 queries x 64 keys per wave) ----
        v8f S[4];
#pragma unroll
        for (int nt = 0; nt < 4; ++nt) {
            v8f sa = vz;
#pragma unroll
            for (int s = 0; s < 2; ++s) {
                AFu bb;
                const __bf16* bp = &Ks[(nt * 16 + nloc) * LDT + s * 32 + half * 8];
                bb.u4[0] = *reinterpret_cast<const uint4*>(bp);
                bb.u4[1] = *reinterpret_cast<const uint4*>(bp + 16);
                sa = __builtin_amdgcn_wmma_f32_16x16x32_bf16(
                    false, qa[s].v, false, bb.v, (short)0, sa, false, false);
            }
            S[nt] = sa;
        }

        // ---- online softmax (row = r + 8*half; its 16 cols live in one
        //      16-lane half => xor-shuffle tree over masks 8..1) ----
#pragma unroll
        for (int r = 0; r < 8; ++r) {
            float mx = fmaxf(fmaxf(S[0][r], S[1][r]), fmaxf(S[2][r], S[3][r]));
#pragma unroll
            for (int m = 8; m > 0; m >>= 1) mx = fmaxf(mx, __shfl_xor(mx, m, 32));
            const float mnew  = fmaxf(mrow[r], mx);
            const float alpha = __expf(mrow[r] - mnew);
            mrow[r] = mnew;
            float psum = 0.f;
#pragma unroll
            for (int nt = 0; nt < 4; ++nt) {
                float p = __expf(S[nt][r] - mnew);
                S[nt][r] = p;
                psum += p;
                Oacc[nt][r] = Oacc[nt][r] * alpha;
            }
#pragma unroll
            for (int m = 8; m > 0; m >>= 1) psum += __shfl_xor(psum, m, 32);
            lrow[r] = lrow[r] * alpha + psum;
        }

        // ---- stage P (C-layout -> LDS row-major -> A-layout) ----
#pragma unroll
        for (int nt = 0; nt < 4; ++nt)
#pragma unroll
            for (int r = 0; r < 8; ++r)
                Pw[(r + 8 * half) * LDT + nt * 16 + nloc] = f2bf(S[nt][r]);
        __syncthreads();

        // ---- O += P * V ----
#pragma unroll
        for (int s = 0; s < 2; ++s) {
            AFu pa;
            const __bf16* pp = &Pw[nloc * LDT + s * 32 + half * 8];
            pa.u4[0] = *reinterpret_cast<const uint4*>(pp);
            pa.u4[1] = *reinterpret_cast<const uint4*>(pp + 16);
#pragma unroll
            for (int nt = 0; nt < 4; ++nt) {
                AFu vb;
                const __bf16* vp = &Vt[(nt * 16 + nloc) * LDT + s * 32 + half * 8];
                vb.u4[0] = *reinterpret_cast<const uint4*>(vp);
                vb.u4[1] = *reinterpret_cast<const uint4*>(vp + 16);
                Oacc[nt] = __builtin_amdgcn_wmma_f32_16x16x32_bf16(
                    false, pa.v, false, vb.v, (short)0, Oacc[nt], false, false);
            }
        }
        __syncthreads();
    }

    // ---- finalize: O / l ----
#pragma unroll
    for (int r = 0; r < 8; ++r) {
        const float inv = 1.0f / lrow[r];
        const size_t rowg = rowbase + qt * 64 + wave * 16 + r + 8 * half;
#pragma unroll
        for (int nt = 0; nt < 4; ++nt)
            o[rowg * DIMM + hc + nt * 16 + nloc] = f2bf(Oacc[nt][r] * inv);
    }
}

// ---------------------------------------------------------------------------
__global__ __launch_bounds__(256) void copy_f32x4(const float* __restrict__ src,
                                                  float* __restrict__ dst) {
    int i = blockIdx.x * 256 + threadIdx.x;
    reinterpret_cast<float4*>(dst)[i] = reinterpret_cast<const float4*>(src)[i];
}

// ---------------------------------------------------------------------------
extern "C" void kernel_launch(void* const* d_in, const int* in_sizes, int n_in,
                              void* d_out, int out_size, void* d_ws, size_t ws_size,
                              hipStream_t stream) {
    const float* x      = (const float*)d_in[0];
    const float* ln1_w  = (const float*)d_in[1];
    const float* ln1_b  = (const float*)d_in[2];
    const float* qkv_w  = (const float*)d_in[3];
    const float* out_w  = (const float*)d_in[4];
    const float* out_b  = (const float*)d_in[5];
    const float* ln2_w  = (const float*)d_in[6];
    const float* ln2_b  = (const float*)d_in[7];
    const float* ff1_w  = (const float*)d_in[8];
    const float* ff1_b  = (const float*)d_in[9];
    const float* ff2_w  = (const float*)d_in[10];
    const float* ff2_b  = (const float*)d_in[11];
    float* out = (float*)d_out;

    // workspace carve-out (~154 MB: 69 MB activations + 85 MB bf16 weights)
    char* w = (char*)d_ws;
    float*  xf = (float*)w;  w += (size_t)RTOK * DIMM * 4;
    __bf16* hb = (__bf16*)w; w += (size_t)RTOK * DIMM * 2;
    __bf16* qb = (__bf16*)w; w += (size_t)RTOK * DIMM * 2;
    __bf16* kb = (__bf16*)w; w += (size_t)RTOK * DIMM * 2;
    __bf16* vb = (__bf16*)w; w += (size_t)RTOK * DIMM * 2;
    __bf16* ob = (__bf16*)w; w += (size_t)RTOK * DIMM * 2;
    __bf16* mb = (__bf16*)w; w += (size_t)RTOK * MLPD * 2;
    __bf16* wt_qkv = (__bf16*)w; w += (size_t)NLAYER * DIMM * 3 * DIMM * 2;
    __bf16* wt_out = (__bf16*)w; w += (size_t)NLAYER * DIMM * DIMM * 2;
    __bf16* wt_ff1 = (__bf16*)w; w += (size_t)NLAYER * DIMM * MLPD * 2;
    __bf16* wt_ff2 = (__bf16*)w; w += (size_t)NLAYER * MLPD * DIMM * 2;

    const float qscale = 1.0f / sqrtf((float)DIMM);   // dim**-0.5 per reference

    // one-time weight convert+transpose: [K x N] f32 -> [N x K] bf16
    wconv_kernel<<<dim3(3 * DIMM / 32, DIMM / 32, NLAYER), 256, 0, stream>>>(qkv_w, wt_qkv, DIMM, 3 * DIMM);
    wconv_kernel<<<dim3(DIMM / 32,     DIMM / 32, NLAYER), 256, 0, stream>>>(out_w, wt_out, DIMM, DIMM);
    wconv_kernel<<<dim3(MLPD / 32,     DIMM / 32, NLAYER), 256, 0, stream>>>(ff1_w, wt_ff1, DIMM, MLPD);
    wconv_kernel<<<dim3(DIMM / 32,     MLPD / 32, NLAYER), 256, 0, stream>>>(ff2_w, wt_ff2, MLPD, DIMM);

    // x -> residual stream
    copy_f32x4<<<RTOK * DIMM / 4 / 256, 256, 0, stream>>>(x, xf);

    const dim3 lnGrid(RTOK / 8);
    const dim3 gQKV(3 * DIMM / 128, RTOK / 128);
    const dim3 gOUT(DIMM / 128,     RTOK / 128);
    const dim3 gFF1(MLPD / 128,     RTOK / 128);
    const dim3 gFF2(DIMM / 128,     RTOK / 128);
    const dim3 gATT(SEQ / 64, NHEAD, NBATCH);

    for (int l = 0; l < NLAYER; ++l) {
        // --- attention block ---
        ln_kernel<<<lnGrid, 256, 0, stream>>>(xf, ln1_w + l * DIMM, ln1_b + l * DIMM, hb);
        gemm_bf16_kernel<0><<<gQKV, 256, 0, stream>>>(
            hb, wt_qkv + (size_t)l * DIMM * 3 * DIMM, nullptr, DIMM, 3 * DIMM,
            qscale, qb, kb, vb, nullptr, nullptr);
        attn_kernel<<<gATT, 128, 0, stream>>>(qb, kb, vb, ob);
        gemm_bf16_kernel<1><<<gOUT, 256, 0, stream>>>(
            ob, wt_out + (size_t)l * DIMM * DIMM, out_b + l * DIMM, DIMM, DIMM,
            1.0f, nullptr, nullptr, nullptr, xf, nullptr);
        // --- feed-forward block ---
        ln_kernel<<<lnGrid, 256, 0, stream>>>(xf, ln2_w + l * DIMM, ln2_b + l * DIMM, hb);
        gemm_bf16_kernel<2><<<gFF1, 256, 0, stream>>>(
            hb, wt_ff1 + (size_t)l * DIMM * MLPD, ff1_b + l * MLPD, DIMM, MLPD,
            1.0f, nullptr, nullptr, nullptr, nullptr, mb);
        gemm_bf16_kernel<1><<<gFF2, 256, 0, stream>>>(
            mb, wt_ff2 + (size_t)l * MLPD * DIMM, ff2_b + l * DIMM, MLPD, DIMM,
            1.0f, nullptr, nullptr, nullptr, xf, nullptr);
    }

    copy_f32x4<<<RTOK * DIMM / 4 / 256, 256, 0, stream>>>(xf, out);
}
